// Head_40922448396370
// MI455X (gfx1250) — matrix operand order repaired and checked by reading, not compiled, not run on hardware
//
#include <hip/hip_runtime.h>

// ---------------------------------------------------------------------------
// Fused attention head for MI455X (gfx1250):
//   bf16 WMMA + flash attention; TDM double-buffered staging of weight tiles
//   (projection) and K/V chunks (attention) into padded LDS.
// ---------------------------------------------------------------------------

typedef __attribute__((ext_vector_type(4)))  unsigned int   v4u;
typedef __attribute__((ext_vector_type(4)))  int            v4i;
typedef __attribute__((ext_vector_type(8)))  int            v8i;
typedef __attribute__((ext_vector_type(8)))  float          v8f;
typedef __attribute__((ext_vector_type(8)))  float          v8fv;
typedef __attribute__((ext_vector_type(8)))  unsigned short v8u16;
typedef __attribute__((ext_vector_type(16))) unsigned short v16u16;
typedef __attribute__((ext_vector_type(16))) __bf16         v16bf;

#if defined(__has_builtin)
#if __has_builtin(__builtin_amdgcn_tensor_load_to_lds)
#define HAVE_TDM 1
#if __has_include(<hip/amd_detail/amd_gfx1250_TDM.h>)
#define TDM_ARITY6 1
#endif
#endif
#endif

__device__ __forceinline__ unsigned short f2bf(float f) {
    unsigned int u = __float_as_uint(f);
    u += 0x7fffu + ((u >> 16) & 1u);          // round-to-nearest-even
    return (unsigned short)(u >> 16);
}

__device__ __forceinline__ v16bf asbf(v16u16 v) {
    v16bf r; __builtin_memcpy(&r, &v, sizeof(r)); return r;
}

__device__ __forceinline__ v16u16 cat8(v8u16 a, v8u16 b) {
    return __builtin_shufflevector(a, b, 0,1,2,3,4,5,6,7,8,9,10,11,12,13,14,15);
}

__device__ __forceinline__ v16u16 ldB(const unsigned short* p) {
    return cat8(*(const v8u16*)p, *(const v8u16*)(p + 8));
}

static constexpr int B  = 4;
static constexpr int T  = 4096;
static constexpr int C  = 2048;
static constexpr int H  = 128;
static constexpr int BT = B * T;
static constexpr float SCALE   = 0.022097086912079608f; // C^-0.5
static constexpr float NEG_BIG = -1e30f;

// LDS row strides after TDM padding:
//  - 32-elem (64B) rows padded +16B -> 80B  (pad_interval=16dw, amount=4dw)
//  - 128-elem (256B) rows padded +16B -> 272B (pad_interval=64dw, amount=4dw)
// Both keep 16B alignment and give conflict-free ds_load_b128 column access.
static constexpr int WROW = 80;
static constexpr int KROW = 272;
static constexpr unsigned TDM_FLAGS_W = 0x06D10000u; // 2B | pad | 16dw | 4dw
static constexpr unsigned TDM_FLAGS_K = 0x07510000u; // 2B | pad | 64dw | 4dw

// ---------------------------------------------------------------------------
// TDM: async-load a 2D bf16 tile into LDS (descriptor per ISA §8.3/8.4).
//  tensor: tensor_d0 x tensor_d1 elements, row stride stride0 elements.
//  tile:   tile_d0 (contiguous) x tile_d1 rows starting at gsrc.
// ---------------------------------------------------------------------------
__device__ __forceinline__ void tdm_load_2d(const unsigned short* gsrc,
                                            unsigned lds_byte_off, unsigned flags,
                                            unsigned tensor_d0, unsigned tensor_d1,
                                            unsigned tile_d0, unsigned tile_d1,
                                            unsigned stride0) {
#ifdef HAVE_TDM
    unsigned long long ga = (unsigned long long)(size_t)gsrc;
    v4u g0;
    g0[0] = 1u;                                            // count=1, user mode
    g0[1] = lds_byte_off;                                  // lds_addr
    g0[2] = (unsigned)ga;                                  // global_addr[31:0]
    g0[3] = (unsigned)((ga >> 32) & 0x1FFFFFFull)          // global_addr[56:32]
          | 0x80000000u;                                   // type=2 ("image")
    v8i g1;
    g1[0] = (int)flags;                 // data_size / pad config
    g1[1] = (int)(tensor_d0 << 16);     // tensor_dim0[15:0]
    g1[2] = (int)(tensor_d1 << 16);     // tensor_dim1[15:0]
    g1[3] = (int)(tile_d0 << 16);       // tile_dim0
    g1[4] = (int)tile_d1;               // tile_dim1
    g1[5] = (int)stride0;               // tensor_dim0_stride
    g1[6] = 0;
    g1[7] = 0;
    v4i gz = (v4i)(0);
#ifdef TDM_ARITY6
    __builtin_amdgcn_tensor_load_to_lds(g0, g1, gz, gz, (v8i)(0), 0);
#else
    __builtin_amdgcn_tensor_load_to_lds(g0, g1, gz, gz, 0);
#endif
#endif
}

// ---------------------------------------------------------------------------
// Kernel 1: convert fp32 weights [C][H] -> bf16 W^T [3][H][C]
// ---------------------------------------------------------------------------
__global__ void convert_w_kernel(const float* __restrict__ Wq,
                                 const float* __restrict__ Wk,
                                 const float* __restrict__ Wv,
                                 unsigned short* __restrict__ wT) {
    int idx = blockIdx.x * blockDim.x + threadIdx.x;
    if (idx >= 3 * C * H) return;
    int widx = idx / (C * H);
    int rem  = idx - widx * (C * H);
    int c = rem / H;
    int h = rem - c * H;
    const float* W = (widx == 0) ? Wq : (widx == 1) ? Wk : Wv;
    wT[(size_t)widx * H * C + (size_t)h * C + c] = f2bf(W[(size_t)c * H + h]);
}

// ---------------------------------------------------------------------------
// Kernel 2: fused Q/K/V projection.  One pass over x (read once!), three
// accumulator banks per wave, three TDM-staged weight tiles per k-step.
// 4 waves / block, each wave one 16-row tile.
// ---------------------------------------------------------------------------
__global__ void __launch_bounds__(128)
qkv_proj_kernel(const float* __restrict__ x,
                const unsigned short* __restrict__ wT_all,
                unsigned short* __restrict__ Qb,
                unsigned short* __restrict__ Kb,
                unsigned short* __restrict__ VTb) {
#ifdef HAVE_TDM
    __shared__ __align__(16) unsigned char wlds[2][3][WROW * H]; // 60KB
#endif
    const int wave = threadIdx.x >> 5;
    const int lane = threadIdx.x & 31;
    const int half = lane >> 4;
    const int l16  = lane & 15;

    const int r0  = (blockIdx.x * 4 + wave) * 16;   // global flat row (b*T + t)
    const int row = r0 + l16;

    v8f acc[3][8];
#pragma unroll
    for (int w = 0; w < 3; ++w)
#pragma unroll
        for (int nt = 0; nt < 8; ++nt) acc[w][nt] = (v8f)(0.0f);

#ifdef HAVE_TDM
    if (wave == 0) {
#pragma unroll
        for (int w = 0; w < 3; ++w)
            tdm_load_2d(wT_all + (size_t)w * H * C,
                        (unsigned)(size_t)&wlds[0][w][0],
                        TDM_FLAGS_W, C, H, 32, H, C);
        __builtin_amdgcn_s_wait_tensorcnt(0);
    }
    __syncthreads();
#endif

    constexpr int NIT = C / 32;
    for (int it = 0; it < NIT; ++it) {
        const int c0  = it * 32;
#ifdef HAVE_TDM
        const int buf = it & 1;
        if (wave == 0 && it + 1 < NIT) {
#pragma unroll
            for (int w = 0; w < 3; ++w)
                tdm_load_2d(wT_all + (size_t)w * H * C + (c0 + 32),
                            (unsigned)(size_t)&wlds[1 - buf][w][0],
                            TDM_FLAGS_W, C, H, 32, H, C);
        }
#endif
        // A fragment: x rows (fp32 -> bf16 on the fly); shared by Q, K, V.
        const float* xp = x + (size_t)row * C + c0 + half * 16;
        __builtin_prefetch(xp + 64, 0, 3);
        v8fv g0 = *(const v8fv*)xp;
        v8fv g1 = *(const v8fv*)(xp + 8);
        v16u16 a;
#pragma unroll
        for (int i = 0; i < 8; ++i) { a[i] = f2bf(g0[i]); a[i + 8] = f2bf(g1[i]); }
        v16bf ab = asbf(a);

#ifdef HAVE_TDM
#pragma unroll
        for (int w = 0; w < 3; ++w) {
            const unsigned char* wb = &wlds[buf][w][0];
            v16u16 bcur = ldB((const unsigned short*)(wb + (size_t)l16 * WROW + half * 32));
#pragma unroll
            for (int nt = 0; nt < 8; ++nt) {
                v16u16 bnext;
                if (nt < 7)
                    bnext = ldB((const unsigned short*)(wb +
                                (size_t)((nt + 1) * 16 + l16) * WROW + half * 32));
                acc[w][nt] = __builtin_amdgcn_wmma_f32_16x16x32_bf16(
                    false, ab, false, asbf(bcur), (short)0, acc[w][nt], false, false);
                bcur = bnext;
            }
        }
        if (wave == 0 && it + 1 < NIT) __builtin_amdgcn_s_wait_tensorcnt(0);
        __syncthreads();
#else
#pragma unroll
        for (int w = 0; w < 3; ++w) {
            const unsigned short* wT = wT_all + (size_t)w * H * C;
            v16u16 bcur = ldB(wT + (size_t)l16 * C + c0 + half * 16);
#pragma unroll
            for (int nt = 0; nt < 8; ++nt) {
                v16u16 bnext;
                if (nt < 7)
                    bnext = ldB(wT + (size_t)((nt + 1) * 16 + l16) * C + c0 + half * 16);
                acc[w][nt] = __builtin_amdgcn_wmma_f32_16x16x32_bf16(
                    false, ab, false, asbf(bcur), (short)0, acc[w][nt], false, false);
                bcur = bnext;
            }
        }
#endif
    }

    // ---- stores ----
    {
        const int b    = r0 / T;
        const int tloc = r0 - b * T;
#pragma unroll
        for (int nt = 0; nt < 8; ++nt) {               // V transposed [B][H][T]
            const int h = nt * 16 + l16;
            v8u16 pk;
#pragma unroll
            for (int j = 0; j < 8; ++j) pk[j] = f2bf(acc[2][nt][j]);
            *(v8u16*)(VTb + (size_t)(b * H + h) * T + tloc + half * 8) = pk;
        }
    }
#pragma unroll
    for (int nt = 0; nt < 8; ++nt) {                   // Q, K row-major
#pragma unroll
        for (int j = 0; j < 8; ++j) {
            const int m = j + 8 * half;
            Qb[(size_t)(r0 + m) * H + nt * 16 + l16] = f2bf(acc[0][nt][j]);
            Kb[(size_t)(r0 + m) * H + nt * 16 + l16] = f2bf(acc[1][nt][j]);
        }
    }
}

// ---------------------------------------------------------------------------
// Kernel 3: flash attention.  Block owns 64 query rows (4 waves x 16 rows);
// K and V^T chunks are TDM-loaded into LDS once per block, double-buffered.
// ---------------------------------------------------------------------------
__global__ void __launch_bounds__(128)
flash_attn_kernel(const unsigned short* __restrict__ Qb,
                  const unsigned short* __restrict__ Kb,
                  const unsigned short* __restrict__ VTb,
                  float* __restrict__ out) {
#ifdef HAVE_TDM
    __shared__ __align__(16) unsigned char klds[2][32 * KROW];   // 17KB
    __shared__ __align__(16) unsigned char vlds[2][WROW * H];    // 20KB
#endif
    __shared__ __align__(16) unsigned char ldsP[4][16 * WROW];   // P bounce, 5KB

    const int wave = threadIdx.x >> 5;
    const int lane = threadIdx.x & 31;
    const int half = lane >> 4;
    const int l16  = lane & 15;

    const int rb0 = blockIdx.x * 64;               // block's first flat row
    const int b   = rb0 / T;
    const int qb0 = rb0 - b * T;                   // block's first row in batch
    const int r0  = rb0 + wave * 16;               // this wave's rows
    const int qt0 = qb0 + wave * 16;

    const unsigned short* Kbase = Kb  + (size_t)b * T * H;
    const unsigned short* Vbase = VTb + (size_t)b * H * T;

    // Full 16x128 Q tile as 4 A-fragments held in registers.
    v16u16 qf[4];
#pragma unroll
    for (int hs = 0; hs < 4; ++hs) {
        const unsigned short* qp = Qb + (size_t)(r0 + l16) * H + hs * 32 + half * 16;
        qf[hs] = cat8(*(const v8u16*)qp, *(const v8u16*)(qp + 8));
    }

    v8f o[8];
#pragma unroll
    for (int nt = 0; nt < 8; ++nt) o[nt] = (v8f)(0.0f);
    float mrow[8], lrow[8];
#pragma unroll
    for (int j = 0; j < 8; ++j) { mrow[j] = -3.0e38f; lrow[j] = 0.0f; }

    const int nchunks = (qb0 + 64) >> 5;           // uniform across the block

#ifdef HAVE_TDM
    if (wave == 0) {
        tdm_load_2d(Kbase, (unsigned)(size_t)&klds[0][0],
                    TDM_FLAGS_K, H, T, H, 32, H);
        tdm_load_2d(Vbase, (unsigned)(size_t)&vlds[0][0],
                    TDM_FLAGS_W, T, H, 32, H, T);
        __builtin_amdgcn_s_wait_tensorcnt(0);
    }
    __syncthreads();
#endif

    for (int ch = 0; ch < nchunks; ++ch) {
        const int kv0 = ch * 32;
#ifdef HAVE_TDM
        const int buf = ch & 1;
        if (wave == 0 && ch + 1 < nchunks) {
            tdm_load_2d(Kbase + (size_t)(kv0 + 32) * H,
                        (unsigned)(size_t)&klds[1 - buf][0],
                        TDM_FLAGS_K, H, T, H, 32, H);
            tdm_load_2d(Vbase + (kv0 + 32),
                        (unsigned)(size_t)&vlds[1 - buf][0],
                        TDM_FLAGS_W, T, H, 32, H, T);
        }
#endif
        // Waves whose rows all precede this chunk skip the math (keep barriers).
        if (kv0 <= qt0 + 15) {
            // ---- S = Q*K^T ----
            v8f s0 = (v8f)(0.0f), s1 = (v8f)(0.0f);
#ifdef HAVE_TDM
            const unsigned char* kb = &klds[buf][0];
#pragma unroll
            for (int hs = 0; hs < 4; ++hs) {
                v16u16 kf0 = ldB((const unsigned short*)(kb +
                                 (size_t)l16 * KROW + hs * 64 + half * 32));
                v16u16 kf1 = ldB((const unsigned short*)(kb +
                                 (size_t)(16 + l16) * KROW + hs * 64 + half * 32));
                s0 = __builtin_amdgcn_wmma_f32_16x16x32_bf16(false, asbf(qf[hs]),
                         false, asbf(kf0), (short)0, s0, false, false);
                s1 = __builtin_amdgcn_wmma_f32_16x16x32_bf16(false, asbf(qf[hs]),
                         false, asbf(kf1), (short)0, s1, false, false);
            }
#else
            const unsigned short* kcol0 = Kbase + (size_t)(kv0 + l16) * H + half * 16;
            const unsigned short* kcol1 = Kbase + (size_t)(kv0 + 16 + l16) * H + half * 16;
#pragma unroll
            for (int hs = 0; hs < 4; ++hs) {
                v16u16 kf0 = ldB(kcol0 + hs * 32);
                v16u16 kf1 = ldB(kcol1 + hs * 32);
                s0 = __builtin_amdgcn_wmma_f32_16x16x32_bf16(false, asbf(qf[hs]),
                         false, asbf(kf0), (short)0, s0, false, false);
                s1 = __builtin_amdgcn_wmma_f32_16x16x32_bf16(false, asbf(qf[hs]),
                         false, asbf(kf1), (short)0, s1, false, false);
            }
#endif
            // ---- scale + causal mask ----
            const int col0 = kv0 + l16;
            const int col1 = col0 + 16;
#pragma unroll
            for (int j = 0; j < 8; ++j) {
                const int rq = qt0 + j + 8 * half;
                float a0 = s0[j] * SCALE;
                float a1 = s1[j] * SCALE;
                s0[j] = (col0 <= rq) ? a0 : NEG_BIG;
                s1[j] = (col1 <= rq) ? a1 : NEG_BIG;
            }

            // ---- online softmax (row reduction over 16 lanes per half-wave) ----
            float alpha[8];
#pragma unroll
            for (int j = 0; j < 8; ++j) {
                float m = fmaxf(s0[j], s1[j]);
#pragma unroll
                for (int off = 1; off < 16; off <<= 1)
                    m = fmaxf(m, __shfl_xor(m, off, 32));
                float mn = fmaxf(mrow[j], m);
                alpha[j] = __expf(mrow[j] - mn);
                mrow[j]  = mn;
                s0[j] = __expf(s0[j] - mn);
                s1[j] = __expf(s1[j] - mn);
                float rs = s0[j] + s1[j];
#pragma unroll
                for (int off = 1; off < 16; off <<= 1)
                    rs += __shfl_xor(rs, off, 32);
                lrow[j] = lrow[j] * alpha[j] + rs;
            }
#pragma unroll
            for (int nt = 0; nt < 8; ++nt)
#pragma unroll
                for (int j = 0; j < 8; ++j) o[nt][j] *= alpha[j];

            // ---- P (C-layout) -> LDS (padded rows) -> A-fragment layout ----
            unsigned char* lw = &ldsP[wave][0];
#pragma unroll
            for (int j = 0; j < 8; ++j) {
                const int m = j + 8 * half;
                *(unsigned short*)(lw + m * WROW + 2 * l16)      = f2bf(s0[j]);
                *(unsigned short*)(lw + m * WROW + 2 * l16 + 32) = f2bf(s1[j]);
            }
            asm volatile("s_wait_dscnt 0x0" ::: "memory");  // same-wave cross-lane RAW

            const unsigned char* lr = lw + (size_t)l16 * WROW;
            v16u16 pf = cat8(*(const v8u16*)(lr + 32 * half),
                             *(const v8u16*)(lr + 16 + 32 * half));
            asm volatile("" ::: "memory");
            v16bf pbf = asbf(pf);

            // ---- O += P * V ----
#ifdef HAVE_TDM
            const unsigned char* vb = &vlds[buf][0];
#pragma unroll
            for (int nt = 0; nt < 8; ++nt) {
                v16u16 vf = ldB((const unsigned short*)(vb +
                                (size_t)(nt * 16 + l16) * WROW + half * 32));
                o[nt] = __builtin_amdgcn_wmma_f32_16x16x32_bf16(
                    false, pbf, false, asbf(vf), (short)0, o[nt], false, false);
            }
#else
#pragma unroll
            for (int nt = 0; nt < 8; ++nt) {
                v16u16 vf = ldB(Vbase + (size_t)(nt * 16 + l16) * T + kv0 + half * 16);
                o[nt] = __builtin_amdgcn_wmma_f32_16x16x32_bf16(
                    false, pbf, false, asbf(vf), (short)0, o[nt], false, false);
            }
#endif
        }

#ifdef HAVE_TDM
        if (wave == 0 && ch + 1 < nchunks) __builtin_amdgcn_s_wait_tensorcnt(0);
        __syncthreads();
#endif
    }

    // ---- normalize and write fp32 output ----
    float inv[8];
#pragma unroll
    for (int j = 0; j < 8; ++j) inv[j] = 1.0f / lrow[j];
#pragma unroll
    for (int nt = 0; nt < 8; ++nt)
#pragma unroll
        for (int j = 0; j < 8; ++j) {
            const int m = j + 8 * half;
            out[(size_t)(r0 + m) * H + nt * 16 + l16] = o[nt][j] * inv[j];
        }
}

// ---------------------------------------------------------------------------
// Host launcher
// ---------------------------------------------------------------------------
extern "C" void kernel_launch(void* const* d_in, const int* in_sizes, int n_in,
                              void* d_out, int out_size, void* d_ws, size_t ws_size,
                              hipStream_t stream) {
    const float* x  = (const float*)d_in[0];
    const float* Wq = (const float*)d_in[1];
    const float* Wk = (const float*)d_in[2];
    const float* Wv = (const float*)d_in[3];

    char* ws = (char*)d_ws;
    unsigned short* wT  = (unsigned short*)ws;                       // 3*H*C bf16
    unsigned short* Qb  = (unsigned short*)(ws + (size_t)3 * H * C * 2);
    unsigned short* Kb  = Qb + (size_t)BT * H;
    unsigned short* VTb = Kb + (size_t)BT * H;

    convert_w_kernel<<<(3 * C * H + 255) / 256, 256, 0, stream>>>(Wq, Wk, Wv, wT);
    qkv_proj_kernel<<<dim3(BT / 64), 128, 0, stream>>>(x, wT, Qb, Kb, VTb);
    flash_attn_kernel<<<dim3(BT / 64), 128, 0, stream>>>(Qb, Kb, VTb,
                                                         (float*)d_out);
}
